// SAMSNLMFilter_58162447122723
// MI455X (gfx1250) — matrix operand order repaired and checked by reading, not compiled, not run on hardware
//
#include <hip/hip_runtime.h>
#include <hip/hip_bf16.h>
#include <math.h>

typedef __bf16 bf16;
typedef __attribute__((ext_vector_type(16))) __bf16 v16bf;
typedef __attribute__((ext_vector_type(8)))  __bf16 v8bf;
typedef __attribute__((ext_vector_type(8)))  float  v8f;
typedef __attribute__((address_space(3)))    __bf16 lds_bf16;

#define BATCH 2
#define CH    64
#define HH    96
#define WW    96
#define NTOK  (HH*WW)          // 9216
#define INNER 3
#define NDIL  3
#define K54   (INNER*NDIL*6)   // 54
#define SM_SCALE 0.125f        // 1/sqrt(64)
#define KTILE 32
#define QW    16               // query rows per wave
#define WPB   4                // waves per block
#define QB    (QW*WPB)         // 64 query rows per block
#define KROW  72               // kbuf row stride (elems): 64 data + 8 pad (144B)
#define VROW  40               // vbuf row stride (elems): 32 data + 8 pad (80B)

// ---------------- workspace layout (bytes, 256-aligned) ----------------
#define XR_OFF   0u                                   // B*3*N f32   = 221184 B
#define CO_OFF   221184u                              // 3*B*3*N f32 = 663552 B
#define TOK_OFF  884736u                              // B*N*C bf16  = 2359296 B
#define TOKT_OFF 3244032u                             // B*C*N bf16  = 2359296 B

// ===================== 1x1 reduce: x[B,64,H,W] -> xr[B,3,H,W] =====================
__global__ void reduce1x1_kernel(const float* __restrict__ x,
                                 const float* __restrict__ w_red,
                                 const float* __restrict__ b_red,
                                 float* __restrict__ xr) {
  int t = blockIdx.x*blockDim.x + threadIdx.x;
  if (t >= BATCH*INNER*NTOK) return;
  int n = t % NTOK;
  int i = (t / NTOK) % INNER;
  int b = t / (NTOK*INNER);
  const float* xp = x + (size_t)b*CH*NTOK + n;
  float acc = b_red[i];
  #pragma unroll 8
  for (int ci = 0; ci < CH; ++ci)
    acc = fmaf(xp[(size_t)ci*NTOK], w_red[i*CH + ci], acc);
  xr[t] = acc;
}

// ============ dilated 3x3 convs: xr -> co[d][B,3,H,W], d = dil-1 in {0,1,2} ============
__global__ void dilconv_kernel(const float* __restrict__ xr,
                               const float* __restrict__ w_dil,
                               const float* __restrict__ b_dil,
                               float* __restrict__ co) {
  int t = blockIdx.x*blockDim.x + threadIdx.x;
  if (t >= NDIL*BATCH*INNER*NTOK) return;
  int w = t % WW;
  int h = (t / WW) % HH;
  int i = (t / NTOK) % INNER;
  int b = (t / (NTOK*INNER)) % BATCH;
  int d = t / (NTOK*INNER*BATCH);
  int dil = d + 1;
  float acc = b_dil[d*INNER + i];
  for (int ci = 0; ci < INNER; ++ci) {
    const float* xp = xr + (size_t)(b*INNER + ci)*NTOK;
    const float* wp = w_dil + ((d*INNER + i)*INNER + ci)*9;
    #pragma unroll
    for (int kh = 0; kh < 3; ++kh) {
      int hh2 = h + (kh-1)*dil;
      if ((unsigned)hh2 >= (unsigned)HH) continue;
      #pragma unroll
      for (int kw = 0; kw < 3; ++kw) {
        int ww2 = w + (kw-1)*dil;
        if ((unsigned)ww2 >= (unsigned)WW) continue;
        acc = fmaf(xp[hh2*WW + ww2], wp[kh*3 + kw], acc);
      }
    }
  }
  co[t] = acc;
}

// ======= C4 transforms + products + 1x1 fuse, LDS-tiled -> tokens (bf16, 2 layouts) =======
// One block per (b, 64 spatial positions): stage w_fuse + cat block in LDS.
__global__ __launch_bounds__(256) void fuse_kernel(const float* __restrict__ xr,
                                                   const float* __restrict__ co,
                                                   const float* __restrict__ w_fuse,
                                                   const float* __restrict__ b_fuse,
                                                   bf16* __restrict__ tok,
                                                   bf16* __restrict__ tokT) {
  __shared__ float wfs[CH][K54 + 1];    // stride 55 (odd) -> conflict-free over c
  __shared__ float cats[QB][K54 + 2];   // stride 56
  const int b  = blockIdx.x / (NTOK/QB);
  const int n0 = (blockIdx.x % (NTOK/QB)) * QB;
  const int tid = threadIdx.x;

  for (int idx = tid; idx < CH*K54; idx += 256)
    wfs[idx / K54][idx % K54] = w_fuse[idx];

  for (int idx = tid; idx < QB*K54; idx += 256) {
    int p = idx / K54, k = idx % K54;
    int n = n0 + p;
    int h = n / WW, w = n % WW;
    int dd = k / 18, jj = (k / 3) % 6, ci = k % 3;
    int hr, wr;
    switch (jj) {                       // identity, flipW, flipH, rot90 k=1,2,3
      case 0:  hr = h;       wr = w;       break;
      case 1:  hr = h;       wr = WW-1-w;  break;
      case 2:  hr = HH-1-h;  wr = w;       break;
      case 3:  hr = w;       wr = WW-1-h;  break;
      case 4:  hr = HH-1-h;  wr = WW-1-w;  break;
      default: hr = HH-1-w;  wr = h;       break;
    }
    float covl = co[(size_t)((dd*BATCH + b)*INNER + ci)*NTOK + n];
    float tf   = xr[(size_t)(b*INNER + ci)*NTOK + hr*WW + wr];
    cats[p][k] = covl * tf;
  }
  __syncthreads();

  const int c  = tid & 63;
  const int p0 = tid >> 6;
  float bias = b_fuse[c];
  for (int p = p0; p < QB; p += 4) {
    float acc = bias;
    #pragma unroll 6
    for (int k = 0; k < K54; ++k) acc = fmaf(wfs[c][k], cats[p][k], acc);
    int n = n0 + p;
    bf16 v = (bf16)acc;
    tok [((size_t)b*NTOK + n)*CH + c]  = v;
    tokT[((size_t)b*CH  + c)*NTOK + n] = v;
  }
}

// ===================== flash attention + residual epilogue =====================
// out = softmax(T T^t / 8) T over tokens T [B,9216,64]; fused: d_out = x + 0.2*out (NCHW)
__device__ __forceinline__ void async_b128(bf16* lp, const bf16* gp) {
  unsigned loff = (unsigned)(unsigned long long)(lds_bf16*)lp;   // LDS byte offset (AS3)
  asm volatile("global_load_async_to_lds_b128 %0, %1, off" :: "v"(loff), "v"(gp) : "memory");
}

__device__ __forceinline__ v16bf ld_frag_lds(const bf16* p) {    // 32B, 16B-aligned halves
  v8bf lo = *(const v8bf*)p;
  v8bf hi = *(const v8bf*)(p + 8);
  v16bf r;
  #pragma unroll
  for (int e = 0; e < 8; ++e) { r[e] = lo[e]; r[e+8] = hi[e]; }
  return r;
}

__global__ __launch_bounds__(WPB*32) void attn_kernel(const bf16* __restrict__ tok,
                                                      const bf16* __restrict__ tokT,
                                                      const float* __restrict__ x,
                                                      float* __restrict__ out) {
  __shared__ __align__(16) bf16 kbuf[2][KTILE][KROW];   // keys row-major, padded rows
  __shared__ __align__(16) bf16 vbuf[2][CH][VROW];      // channel-major, padded rows
  __shared__ bf16 pbuf[WPB][16][33];                    // per-wave P-transpose buffer

  const int tid  = threadIdx.x;
  const int lane = tid & 31;
  const int wave = tid >> 5;
  const int half = lane >> 4;
  const int l16  = lane & 15;
  const int nblk = NTOK / QB;                           // 144
  const int b    = blockIdx.x / nblk;
  const int qbase = (blockIdx.x % nblk)*QB + wave*QW;

  const bf16* tb  = tok  + (size_t)b*NTOK*CH;
  const bf16* tbT = tokT + (size_t)b*CH*NTOK;

  // ---- Q fragments (A layout 16x32: lane = row M, contiguous K: k = 16*half + e) ----
  const bf16* tq = tb + (size_t)(qbase + l16)*CH;
  v16bf a0, a1;
  {
    v8bf q0 = *(const v8bf*)(tq + half*16);
    v8bf q1 = *(const v8bf*)(tq + 8 + half*16);
    v8bf q2 = *(const v8bf*)(tq + 32 + half*16);
    v8bf q3 = *(const v8bf*)(tq + 40 + half*16);
    #pragma unroll
    for (int e = 0; e < 8; ++e) { a0[e]=q0[e]; a0[e+8]=q1[e]; a1[e]=q2[e]; a1[e+8]=q3[e]; }
  }

  v8f o0={}, o1={}, o2={}, o3={};
  v8f mrow, lrow;
  #pragma unroll
  for (int r = 0; r < 8; ++r) { mrow[r] = -3.0e38f; lrow[r] = 0.f; }

  // ---- stage tile 0 asynchronously (4 async b128 per thread per tile) ----
  #define STAGE(buf, kt)                                                         \
    do {                                                                         \
      _Pragma("unroll")                                                          \
      for (int j = 0; j < 2; ++j) {                                              \
        int idx = tid + j*128;           /* 256 x 16B chunks: K tile */          \
        int r = idx >> 3, q = idx & 7;                                           \
        async_b128(&kbuf[buf][r][q*8], tb + (size_t)((kt) + r)*CH + q*8);        \
      }                                                                          \
      _Pragma("unroll")                                                          \
      for (int j = 0; j < 2; ++j) {                                              \
        int idx = tid + j*128;           /* 256 x 16B chunks: V tile */          \
        int r = idx >> 2, q = idx & 3;                                           \
        async_b128(&vbuf[buf][r][q*8], tbT + (size_t)r*NTOK + (kt) + q*8);       \
      }                                                                          \
    } while (0)

  STAGE(0, 0);

  const int ntiles = NTOK / KTILE;                      // 288
  for (int it = 0; it < ntiles; ++it) {
    const int kt  = it * KTILE;
    const int cur = it & 1;
    const int nktw = (kt + KTILE) % NTOK;               // wrap: keeps pipeline uniform
    STAGE(cur ^ 1, nktw);

    __builtin_prefetch(tb + (size_t)((kt + 2*KTILE) % NTOK + lane)*CH, 0, 1);

    // wait for OUR tile-`it` async loads (in-order: only the 4 just-issued remain)
    asm volatile("s_wait_asynccnt 0x4" ::: "memory");
    __syncthreads();                                    // all waves' data landed

    // ---- K^T fragments from LDS (B layout 32x16: lane = key col, 16 contig channels) ----
    v16bf bk00 = ld_frag_lds(&kbuf[cur][     l16][     half*16]);
    v16bf bk01 = ld_frag_lds(&kbuf[cur][     l16][32 + half*16]);
    v16bf bk10 = ld_frag_lds(&kbuf[cur][16 + l16][     half*16]);
    v16bf bk11 = ld_frag_lds(&kbuf[cur][16 + l16][32 + half*16]);
    // ---- V fragments from LDS (B layout 32x16: lane = channel col, 16 contig keys) ----
    v16bf bv0  = ld_frag_lds(&vbuf[cur][ 0 + l16][half*16]);
    v16bf bv1  = ld_frag_lds(&vbuf[cur][16 + l16][half*16]);
    v16bf bv2  = ld_frag_lds(&vbuf[cur][32 + l16][half*16]);
    v16bf bv3  = ld_frag_lds(&vbuf[cur][48 + l16][half*16]);

    v8f s0 = {}, s1 = {};
    s0 = __builtin_amdgcn_wmma_f32_16x16x32_bf16(false, a0, false, bk00, (short)0, s0, false, false);
    s0 = __builtin_amdgcn_wmma_f32_16x16x32_bf16(false, a1, false, bk01, (short)0, s0, false, false);
    s1 = __builtin_amdgcn_wmma_f32_16x16x32_bf16(false, a0, false, bk10, (short)0, s1, false, false);
    s1 = __builtin_amdgcn_wmma_f32_16x16x32_bf16(false, a1, false, bk11, (short)0, s1, false, false);

    // ---- online softmax; element r holds query row (r + half*8), N = l16 ----
    v8f tmax;
    #pragma unroll
    for (int r = 0; r < 8; ++r) {
      s0[r] *= SM_SCALE; s1[r] *= SM_SCALE;
      tmax[r] = fmaxf(s0[r], s1[r]);
    }
    #pragma unroll
    for (int m = 1; m < 16; m <<= 1)
      #pragma unroll
      for (int r = 0; r < 8; ++r)
        tmax[r] = fmaxf(tmax[r], __shfl_xor(tmax[r], m, 32));

    v8f pr0, pr1, rsum;
    #pragma unroll
    for (int r = 0; r < 8; ++r) {
      float mn = fmaxf(mrow[r], tmax[r]);
      float al = __expf(mrow[r] - mn);
      mrow[r] = mn;
      pr0[r] = __expf(s0[r] - mn);
      pr1[r] = __expf(s1[r] - mn);
      rsum[r] = pr0[r] + pr1[r];
      lrow[r] *= al;
      o0[r] *= al; o1[r] *= al; o2[r] *= al; o3[r] *= al;
    }
    #pragma unroll
    for (int m = 1; m < 16; m <<= 1)
      #pragma unroll
      for (int r = 0; r < 8; ++r)
        rsum[r] += __shfl_xor(rsum[r], m, 32);
    #pragma unroll
    for (int r = 0; r < 8; ++r) lrow[r] += rsum[r];

    // ---- transpose P: C/D layout -> A layout through wave-private LDS ----
    #pragma unroll
    for (int r = 0; r < 8; ++r) {
      pbuf[wave][r + half*8][l16]      = (bf16)pr0[r];
      pbuf[wave][r + half*8][l16 + 16] = (bf16)pr1[r];
    }
    asm volatile("s_wait_dscnt 0x0" ::: "memory");
    v16bf pA;
    #pragma unroll
    for (int e = 0; e < 16; ++e) {
      int k = e + half*16;                              // contiguous K: k = 16*half + e
      pA[e] = pbuf[wave][l16][k];
    }

    o0 = __builtin_amdgcn_wmma_f32_16x16x32_bf16(false, pA, false, bv0, (short)0, o0, false, false);
    o1 = __builtin_amdgcn_wmma_f32_16x16x32_bf16(false, pA, false, bv1, (short)0, o1, false, false);
    o2 = __builtin_amdgcn_wmma_f32_16x16x32_bf16(false, pA, false, bv2, (short)0, o2, false, false);
    o3 = __builtin_amdgcn_wmma_f32_16x16x32_bf16(false, pA, false, bv3, (short)0, o3, false, false);

    __syncthreads();                                    // tile consumed: safe to overwrite
  }
  #undef STAGE

  // ---- epilogue: 1/l normalize + residual + [B,N,C] -> [B,C,H,W] ----
  #pragma unroll
  for (int r = 0; r < 8; ++r) {
    float inv = 1.0f / lrow[r];
    int n = qbase + r + half*8;
    #pragma unroll
    for (int cg = 0; cg < 4; ++cg) {
      float v = (cg==0 ? o0[r] : cg==1 ? o1[r] : cg==2 ? o2[r] : o3[r]) * inv;
      int c = cg*16 + l16;
      size_t idx = ((size_t)b*CH + c)*NTOK + n;
      out[idx] = x[idx] + 0.2f * v;
    }
  }
}

// ===================== host launcher =====================
extern "C" void kernel_launch(void* const* d_in, const int* in_sizes, int n_in,
                              void* d_out, int out_size, void* d_ws, size_t ws_size,
                              hipStream_t stream) {
  const float* x      = (const float*)d_in[0];
  const float* w_red  = (const float*)d_in[1];
  const float* b_red  = (const float*)d_in[2];
  const float* w_dil  = (const float*)d_in[3];
  const float* b_dil  = (const float*)d_in[4];
  const float* w_fuse = (const float*)d_in[5];
  const float* b_fuse = (const float*)d_in[6];
  float* out = (float*)d_out;

  char* ws = (char*)d_ws;
  float* xr   = (float*)(ws + XR_OFF);
  float* co   = (float*)(ws + CO_OFF);
  bf16*  tok  = (bf16*) (ws + TOK_OFF);
  bf16*  tokT = (bf16*) (ws + TOKT_OFF);

  reduce1x1_kernel<<<(BATCH*INNER*NTOK + 255)/256, 256, 0, stream>>>(x, w_red, b_red, xr);
  dilconv_kernel  <<<(NDIL*BATCH*INNER*NTOK + 255)/256, 256, 0, stream>>>(xr, w_dil, b_dil, co);
  fuse_kernel     <<<BATCH*(NTOK/QB), 256, 0, stream>>>(xr, co, w_fuse, b_fuse, tok, tokT);
  attn_kernel     <<<BATCH*(NTOK/QB), WPB*32, 0, stream>>>(tok, tokT, x, out);
}
